// Gemma4Attention_55808805044537
// MI455X (gfx1250) — matrix operand order repaired and checked by reading, not compiled, not run on hardware
//
#include <hip/hip_runtime.h>
#include <hip/hip_bf16.h>

// ---------------------------------------------------------------------------
// Gemma4 attention block for MI455X (gfx1250, wave32, WMMA bf16 16x16x32).
//   HID=2048, H=16, KV=8, D=128, T=2048, SW=512
// Pipeline (5 launches):
//   1..3) gemm_fused mode 0/1/2 : q/k/v = X @ W^T, fused RMSNorm (+RoPE for q,k),
//         bf16 output to workspace. fp32->bf16 conversion happens in LDS staging.
//   4) attn_kernel : flash-style sliding-window attention, bf16 WMMA,
//      K/V blocks staged via GLOBAL_LOAD_ASYNC_TO_LDS (ASYNCcnt) when available.
//   5) gemm_fused mode 3 : out = attn @ o_w^T, fp32 output (A staged async).
// ---------------------------------------------------------------------------

typedef __attribute__((ext_vector_type(16))) __bf16 v16bf;
typedef __attribute__((ext_vector_type(8)))  float  v8f;
typedef __attribute__((ext_vector_type(4)))  int    v4i;

#define T_TOK  2048
#define HID_K  2048
#define NH_Q   16
#define NH_KV  8
#define HD     128
#define SWIN   512
#define NEGF   (-1e30f)

// ---- gfx1250 async global->LDS copy (ASYNCcnt path), guarded ----
#if defined(__has_builtin)
#  if __has_builtin(__builtin_amdgcn_global_load_async_to_lds_b128)
#    define HAVE_ASYNC_COPY 1
#  else
#    define HAVE_ASYNC_COPY 0
#  endif
#else
#  define HAVE_ASYNC_COPY 0
#endif

#if HAVE_ASYNC_COPY
typedef __attribute__((address_space(1))) v4i* gv4i_p;   // global int4*
typedef __attribute__((address_space(3))) v4i* lv4i_p;   // LDS int4*
__device__ __forceinline__ void async_cp16(const void* g, void* l) {
    __builtin_amdgcn_global_load_async_to_lds_b128((gv4i_p)g, (lv4i_p)l, 0, 0);
}
__device__ __forceinline__ void async_wait0() {
#if __has_builtin(__builtin_amdgcn_s_wait_asynccnt)
    __builtin_amdgcn_s_wait_asynccnt(0);
#else
    asm volatile("s_wait_asynccnt 0" ::: "memory");   // SOPP opcode 74
#endif
}
#endif

__device__ __forceinline__ v8f vzero8() {
    v8f z;
#pragma unroll
    for (int i = 0; i < 8; ++i) z[i] = 0.f;
    return z;
}

// A fragment: 16x32 bf16, row m = lane&15; contiguous k order, same as the B loaders.
// lane<16: K = {0..15}; lane>=16: K = {16..31}
__device__ __forceinline__ v16bf load_a_frag(const __bf16* base, int ldm, int lane) {
    int m = lane & 15, hi = lane >> 4;
    const __bf16* r = base + m * ldm;
    v16bf a;
#pragma unroll
    for (int p = 0; p < 8; ++p) {
        int k = hi * 16 + p * 2;
        a[2 * p]     = r[k];
        a[2 * p + 1] = r[k + 1];
    }
    return a;
}

// B fragment (32x16) when storage is [n][k] row-major (transposed source, e.g. W rows).
// element e <-> K = e + 16*(lane>=16), N = lane&15  (contiguous k order, matches load_a_frag)
__device__ __forceinline__ v16bf load_b_frag_t(const __bf16* base, int ldn, int lane) {
    int n = lane & 15, hi = lane >> 4;
    const __bf16* r = base + n * ldn + hi * 16;
    v16bf b;
#pragma unroll
    for (int e = 0; e < 16; ++e) b[e] = r[e];
    return b;
}

// B fragment (32x16) when storage is natural [k][n] row-major (e.g. V block).
__device__ __forceinline__ v16bf load_b_frag_n(const __bf16* base, int ldk, int lane) {
    int n = lane & 15, hi = lane >> 4;
    const __bf16* r = base + (hi * 16) * ldk + n;
    v16bf b;
#pragma unroll
    for (int e = 0; e < 16; ++e) b[e] = r[e * ldk];
    return b;
}

// C/D f32 16x16: VGPR p, lane<16 -> (M=p, N=lane); lane>=16 -> (M=p+8, N=lane-16)
__device__ __forceinline__ void store_c(float* base, int ldm, int lane, v8f c) {
    int n = lane & 15, hi = lane >> 4;
#pragma unroll
    for (int p = 0; p < 8; ++p) base[(p + 8 * hi) * ldm + n] = c[p];
}

#define WMMA_BF16(A, B, C) \
    __builtin_amdgcn_wmma_f32_16x16x32_bf16(false, (A), false, (B), (short)0, (C), false, false)

// ---------------------------------------------------------------------------
// Fused GEMM: C[64x128 tile] = A[T,2048] @ W^T, epilogue per mode.
// mode 0: q  (RMSNorm w/ q_norm_w + RoPE, bf16 out, head stride NH_Q)
// mode 1: k  (RMSNorm w/ k_norm_w + RoPE, bf16 out, head stride NH_KV)
// mode 2: v  (plain RMSNorm, bf16 out, head stride NH_KV)
// mode 3: o  (A operand is bf16 attn buffer, plain f32 out)
// grid: (T/64, numHeads or numNTiles), block 256 (8 waves, 4x2 tile of 16x16 wmma)
// ---------------------------------------------------------------------------
__global__ __launch_bounds__(256) void gemm_fused(
    const float*  __restrict__ Af,   // fp32 A (modes 0-2)
    const __bf16* __restrict__ Ab,   // bf16 A (mode 3)
    const float*  __restrict__ W,    // fp32 weights [N,2048], row-major
    const float*  __restrict__ cosp, const float* __restrict__ sinp,
    const float*  __restrict__ normw,
    __bf16* __restrict__ outb, float* __restrict__ outf,
    int nh, int mode)
{
    constexpr int ASTR = 72, BSTR = 72, CSTR = 132;
    __shared__ __align__(16) char smem[64 * CSTR * 4];   // 33792 B
    __bf16* As = (__bf16*)smem;                          //  64x72 bf16 = 9216
    __bf16* Bs = (__bf16*)(smem + 9216);                 // 128x72 bf16 = 18432
    float*  Cs = (float*)smem;                           //  64x132 f32 (aliases staging)
    __shared__ float red[256];

    const int tid  = threadIdx.x;
    const int lane = tid & 31;
    const int w    = tid >> 5;
    const int mw   = w & 3;       // wave row   (4 x 16 = 64 rows)
    const int nw   = w >> 2;      // wave col   (2 x 64 = 128 cols)
    const int r0   = blockIdx.x * 64;
    const int tn   = blockIdx.y;  // head index / n-tile
    const int wrow0 = tn * HD;

    v8f acc[4];
#pragma unroll
    for (int t = 0; t < 4; ++t) acc[t] = vzero8();

    for (int k0 = 0; k0 < HID_K; k0 += 64) {
        // ---- stage A tile 64x64 (convert f32->bf16 unless mode 3) ----
        {
            int row = tid >> 2, cg = tid & 3;       // 16 elems per thread
            if (mode == 3) {
                const __bf16* g = Ab + (size_t)(r0 + row) * HID_K + k0 + cg * 16;
                __bf16* l = As + row * ASTR + cg * 16;
#if HAVE_ASYNC_COPY
                async_cp16(g, l);
                async_cp16(g + 8, l + 8);
#else
                const uint4* s = (const uint4*)g;
                uint4* d = (uint4*)l;
                d[0] = s[0]; d[1] = s[1];
#endif
            } else {
                const float4* s = (const float4*)(Af + (size_t)(r0 + row) * HID_K + k0 + cg * 16);
#pragma unroll
                for (int u = 0; u < 4; ++u) {
                    float4 f = s[u];
                    int o = row * ASTR + cg * 16 + u * 4;
                    As[o]     = (__bf16)f.x; As[o + 1] = (__bf16)f.y;
                    As[o + 2] = (__bf16)f.z; As[o + 3] = (__bf16)f.w;
                }
            }
        }
        // ---- stage B tile 128x64 from W (f32 -> bf16) ----
        {
            int row = tid >> 1, half = tid & 1;     // 32 elems per thread
            const float4* s = (const float4*)(W + (size_t)(wrow0 + row) * HID_K + k0 + half * 32);
#pragma unroll
            for (int u = 0; u < 8; ++u) {
                float4 f = s[u];
                int o = row * BSTR + half * 32 + u * 4;
                Bs[o]     = (__bf16)f.x; Bs[o + 1] = (__bf16)f.y;
                Bs[o + 2] = (__bf16)f.z; Bs[o + 3] = (__bf16)f.w;
            }
        }
#if HAVE_ASYNC_COPY
        if (mode == 3) async_wait0();
#endif
        __syncthreads();

#pragma unroll
        for (int k2 = 0; k2 < 2; ++k2) {
            v16bf af = load_a_frag(As + (mw * 16) * ASTR + k2 * 32, ASTR, lane);
#pragma unroll
            for (int t = 0; t < 4; ++t) {
                v16bf bf = load_b_frag_t(Bs + (nw * 64 + t * 16) * BSTR + k2 * 32, BSTR, lane);
                acc[t] = WMMA_BF16(af, bf, acc[t]);
            }
        }
        __syncthreads();
    }

    // ---- spill C tile to LDS for row-wise epilogue ----
#pragma unroll
    for (int t = 0; t < 4; ++t)
        store_c(Cs + (mw * 16) * CSTR + nw * 64 + t * 16, CSTR, lane, acc[t]);
    __syncthreads();

    const int rid = tid >> 2, qq = tid & 3;     // 4 threads per row, 32 cols each
    float ss = 0.f;
#pragma unroll
    for (int c = 0; c < 32; ++c) {
        float v = Cs[rid * CSTR + qq * 32 + c];
        ss += v * v;
    }
    red[rid * 4 + qq] = ss;
    __syncthreads();
    float mean = (red[rid * 4] + red[rid * 4 + 1] + red[rid * 4 + 2] + red[rid * 4 + 3]) * (1.f / 128.f);
    float sc = rsqrtf(mean + 1e-6f);
    size_t trow = (size_t)(r0 + rid);

    if (mode <= 1) {
        // RMSNorm with weight, in-place, then RoPE (needs both halves)
#pragma unroll
        for (int c = 0; c < 32; ++c) {
            int cc = qq * 32 + c;
            Cs[rid * CSTR + cc] *= sc * normw[cc];
        }
        __syncthreads();
        const float* cr = cosp + trow * HD;
        const float* sr = sinp + trow * HD;
        __bf16* orow = outb + (trow * (size_t)nh + tn) * HD;
#pragma unroll
        for (int c = 0; c < 32; ++c) {
            int cc = qq * 32 + c;
            float y = Cs[rid * CSTR + cc];
            float rot = (cc < 64) ? -Cs[rid * CSTR + cc + 64] : Cs[rid * CSTR + cc - 64];
            orow[cc] = (__bf16)(y * cr[cc] + rot * sr[cc]);
        }
    } else if (mode == 2) {
        __bf16* orow = outb + (trow * (size_t)nh + tn) * HD;
#pragma unroll
        for (int c = 0; c < 32; ++c) {
            int cc = qq * 32 + c;
            orow[cc] = (__bf16)(Cs[rid * CSTR + cc] * sc);
        }
    } else {
        float* orow = outf + trow * 2048 + tn * HD;
#pragma unroll
        for (int c = 0; c < 32; ++c) {
            int cc = qq * 32 + c;
            orow[cc] = Cs[rid * CSTR + cc];
        }
    }
}

__device__ __forceinline__ int seg_of(int i, const int* cu, int ncu) {
    int s = 0;
    for (int c = 0; c < ncu; ++c) s += (i >= cu[c]) ? 1 : 0;   // searchsorted 'right'
    return s;
}

// ---------------------------------------------------------------------------
// Flash-style sliding-window attention. grid (T/64, H), block 256.
// Block = one head, 64 q rows. Online softmax; S and P via bf16 WMMA.
// ---------------------------------------------------------------------------
__global__ __launch_bounds__(256) void attn_kernel(
    const __bf16* __restrict__ qh,  // [T, 16, 128]
    const __bf16* __restrict__ kh,  // [T, 8, 128]
    const __bf16* __restrict__ vh,  // [T, 8, 128]
    __bf16* __restrict__ ab,        // [T, 2048]
    const int* __restrict__ cu, int ncu)
{
    constexpr int KSTR = 136, VSTR = 136, SSTR = 68, PSTR = 72;
    __shared__ __align__(16) __bf16 Ks[64 * KSTR];
    __shared__ __align__(16) __bf16 Vs[64 * VSTR];
    __shared__ __align__(16) float  Sbuf[64 * SSTR];
    __shared__ __align__(16) __bf16 Pbuf[64 * PSTR];
    __shared__ float m_s[64], l_s[64], alpha_s[64];
    __shared__ float redm[256], reds[256];

    const int tid  = threadIdx.x;
    const int lane = tid & 31;
    const int w    = tid >> 5;
    const int mw   = w & 3;       // 4 x 16 q-rows
    const int nw   = w >> 2;      // 2 x 64 output cols / 2 x 32 S cols
    const int hi   = lane >> 4;
    const int r0   = blockIdx.x * 64;
    const int h    = blockIdx.y;
    const int hk   = h >> 1;      // GQA: n_rep = 2

    if (tid < 64) { m_s[tid] = NEGF; l_s[tid] = 0.f; }

    // Q fragments resident for the whole loop: rows mw*16.., D split into 4 k-slices
    v16bf qf[4];
    const __bf16* qbase = qh + ((size_t)(r0 + mw * 16) * NH_Q + h) * HD;
#pragma unroll
    for (int ks = 0; ks < 4; ++ks)
        qf[ks] = load_a_frag(qbase + ks * 32, NH_Q * HD, lane);

    v8f acc[4];
#pragma unroll
    for (int t = 0; t < 4; ++t) acc[t] = vzero8();

    // union window start for rows r0..r0+63 is max(seg_start(r0), r0-SW+1)
    int segStart = 0;
    for (int c = 0; c < ncu; ++c) { int vcu = cu[c]; if (vcu <= r0) segStart = vcu; }
    int jlo = r0 - (SWIN - 1);
    if (segStart > jlo) jlo = segStart;
    if (jlo < 0) jlo = 0;
    const int jb0 = jlo & ~63;

    const int rid = tid >> 2, qq = tid & 3;   // 4 threads/row for softmax bookkeeping
    const int i_row = r0 + rid;
    const int si = seg_of(i_row, cu, ncu);

    for (int jb = jb0; jb <= r0; jb += 64) {
        // ---- stage K/V block [64 x 128] bf16 (async global->LDS if available) ----
        {
            int row = tid >> 2, part = tid & 3;  // 32 elems each
            const __bf16* gk = kh + (((size_t)(jb + row)) * NH_KV + hk) * HD + part * 32;
            const __bf16* gv = vh + (((size_t)(jb + row)) * NH_KV + hk) * HD + part * 32;
            __bf16* lk = Ks + row * KSTR + part * 32;
            __bf16* lv = Vs + row * VSTR + part * 32;
#if HAVE_ASYNC_COPY
#pragma unroll
            for (int u = 0; u < 4; ++u) {
                async_cp16(gk + u * 8, lk + u * 8);
                async_cp16(gv + u * 8, lv + u * 8);
            }
#else
            const uint4* sk = (const uint4*)gk;
            uint4* dk = (uint4*)lk;
            dk[0] = sk[0]; dk[1] = sk[1]; dk[2] = sk[2]; dk[3] = sk[3];
            const uint4* sv = (const uint4*)gv;
            uint4* dv = (uint4*)lv;
            dv[0] = sv[0]; dv[1] = sv[1]; dv[2] = sv[2]; dv[3] = sv[3];
#endif
            // prefetch next block while this one lands (global_prefetch_b8)
            if (jb + 64 <= r0) {
                __builtin_prefetch(gk + (size_t)64 * NH_KV * HD, 0, 1);
                __builtin_prefetch(gv + (size_t)64 * NH_KV * HD, 0, 1);
            }
#if HAVE_ASYNC_COPY
            async_wait0();
#endif
        }
        __syncthreads();

        // ---- S = Q @ K^T : each wave 2 tiles (rows mw*16, cols nw*32 + st*16) ----
#pragma unroll
        for (int st = 0; st < 2; ++st) {
            v8f s = vzero8();
#pragma unroll
            for (int ks = 0; ks < 4; ++ks) {
                v16bf bf = load_b_frag_t(Ks + (nw * 32 + st * 16) * KSTR + ks * 32, KSTR, lane);
                s = WMMA_BF16(qf[ks], bf, s);
            }
            store_c(Sbuf + (mw * 16) * SSTR + nw * 32 + st * 16, SSTR, lane, s);
        }
        __syncthreads();

        // ---- mask + block row-max ----
        float sv[16];
        float lm = NEGF;
#pragma unroll
        for (int c = 0; c < 16; ++c) {
            int j = jb + qq * 16 + c;
            float v = Sbuf[rid * SSTR + qq * 16 + c];
            bool ok = (j <= i_row) && (i_row - j < SWIN) && (seg_of(j, cu, ncu) == si);
            sv[c] = ok ? v : NEGF;
            if (sv[c] > lm) lm = sv[c];
        }
        redm[rid * 4 + qq] = lm;
        __syncthreads();
        if (qq == 0) {
            float bm = redm[rid * 4];
            bm = fmaxf(bm, redm[rid * 4 + 1]);
            bm = fmaxf(bm, redm[rid * 4 + 2]);
            bm = fmaxf(bm, redm[rid * 4 + 3]);
            float mo = m_s[rid];
            float mn = fmaxf(mo, bm);
            alpha_s[rid] = __expf(mo - mn);   // mo==mn -> 1 ; mo=NEG,mn real -> 0
            m_s[rid] = mn;
        }
        __syncthreads();

        // ---- P = exp(S - m_new), bf16 to LDS; partial row sums ----
        float mn = m_s[rid];
        float ps = 0.f;
#pragma unroll
        for (int c = 0; c < 16; ++c) {
            float p = (sv[c] > -1e29f) ? __expf(sv[c] - mn) : 0.f;
            Pbuf[rid * PSTR + qq * 16 + c] = (__bf16)p;
            ps += p;
        }
        reds[rid * 4 + qq] = ps;
        __syncthreads();
        if (qq == 0) {
            l_s[rid] = l_s[rid] * alpha_s[rid]
                     + reds[rid * 4] + reds[rid * 4 + 1] + reds[rid * 4 + 2] + reds[rid * 4 + 3];
        }

        // ---- rescale O accumulators by alpha, then O += P @ V ----
        float al[8];
#pragma unroll
        for (int p = 0; p < 8; ++p) al[p] = alpha_s[mw * 16 + p + 8 * hi];
#pragma unroll
        for (int t = 0; t < 4; ++t)
#pragma unroll
            for (int p = 0; p < 8; ++p) acc[t][p] *= al[p];

#pragma unroll
        for (int j2 = 0; j2 < 2; ++j2) {
            v16bf pf = load_a_frag(Pbuf + (mw * 16) * PSTR + j2 * 32, PSTR, lane);
#pragma unroll
            for (int t = 0; t < 4; ++t) {
                v16bf vf = load_b_frag_n(Vs + (j2 * 32) * VSTR + nw * 64 + t * 16, VSTR, lane);
                acc[t] = WMMA_BF16(pf, vf, acc[t]);
            }
        }
        __syncthreads();   // protects Ks/Vs/Pbuf before next stage; publishes l_s
    }

    // ---- O /= l, store bf16 to attn buffer [T, H*D] ----
    const int nn = lane & 15;
#pragma unroll
    for (int t = 0; t < 4; ++t)
#pragma unroll
        for (int p = 0; p < 8; ++p) {
            int row = mw * 16 + p + 8 * hi;
            float invl = 1.f / l_s[row];
            int col = nw * 64 + t * 16 + nn;
            ab[(size_t)(r0 + row) * 2048 + h * HD + col] = (__bf16)(acc[t][p] * invl);
        }
}

// ---------------------------------------------------------------------------
extern "C" void kernel_launch(void* const* d_in, const int* in_sizes, int n_in,
                              void* d_out, int out_size, void* d_ws, size_t ws_size,
                              hipStream_t stream) {
    (void)n_in; (void)out_size; (void)ws_size;
    const float* x    = (const float*)d_in[0];
    const float* cosp = (const float*)d_in[1];
    const float* sinp = (const float*)d_in[2];
    const float* qw   = (const float*)d_in[3];
    const float* kw   = (const float*)d_in[4];
    const float* vw   = (const float*)d_in[5];
    const float* ow   = (const float*)d_in[6];
    const float* qnw  = (const float*)d_in[7];
    const float* knw  = (const float*)d_in[8];
    const int*   cu   = (const int*)d_in[9];
    const int    ncu  = in_sizes[9];
    float* out = (float*)d_out;

    // workspace layout (bf16): qh 8MiB | kh 4MiB | vh 4MiB | attn 8MiB = 24MiB
    char* ws = (char*)d_ws;
    __bf16* qh = (__bf16*)(ws);
    __bf16* kh = (__bf16*)(ws + ((size_t)8  << 20));
    __bf16* vh = (__bf16*)(ws + ((size_t)12 << 20));
    __bf16* ab = (__bf16*)(ws + ((size_t)16 << 20));

    dim3 blk(256);
    gemm_fused<<<dim3(T_TOK / 64, NH_Q),  blk, 0, stream>>>(x, nullptr, qw, cosp, sinp, qnw, qh, nullptr, NH_Q,  0);
    gemm_fused<<<dim3(T_TOK / 64, NH_KV), blk, 0, stream>>>(x, nullptr, kw, cosp, sinp, knw, kh, nullptr, NH_KV, 1);
    gemm_fused<<<dim3(T_TOK / 64, NH_KV), blk, 0, stream>>>(x, nullptr, vw, nullptr, nullptr, nullptr, vh, nullptr, NH_KV, 2);
    attn_kernel<<<dim3(T_TOK / 64, NH_Q), blk, 0, stream>>>(qh, kh, vh, ab, cu, ncu);
    gemm_fused<<<dim3(T_TOK / 64, NH_Q),  blk, 0, stream>>>(nullptr, ab, ow, nullptr, nullptr, nullptr, nullptr, out, 0, 3);
}
